// MoeModule_19000935318065
// MI455X (gfx1250) — matrix operand order repaired and checked by reading, not compiled, not run on hardware
//
#include <hip/hip_runtime.h>

// Problem constants (from reference): B=4,S=2048,D=1024 -> T=8192; E=8; DFF=4096
// capacity = int(1.25*8192/8)=1280 (even) -> CAP=1280
#define T_TOK 8192
#define DMODEL 1024
#define NEXP 8
#define DFF_ 4096
#define CAP 1280

typedef __attribute__((ext_vector_type(16))) _Float16 v16h;
typedef __attribute__((ext_vector_type(8)))  _Float16 h8;
typedef __attribute__((ext_vector_type(2)))  _Float16 h2;
typedef __attribute__((ext_vector_type(8)))  float    v8f;
typedef __attribute__((ext_vector_type(4)))  float    f4v;

__device__ __forceinline__ float gelu_tanh(float x) {
  const float c0 = 0.7978845608028654f;  // sqrt(2/pi)
  const float c1 = 0.044715f;
  float t = tanhf(c0 * (x + c1 * x * x * x));
  return 0.5f * x * (1.0f + t);
}

// ---------------------------------------------------------------------------
// 1) Gating: logits = tokens @ gw^T, softmax, argmax (first max wins).
//    One wave per token; gate weights staged in LDS. Deterministic prob sums.
// ---------------------------------------------------------------------------
__global__ __launch_bounds__(256) void gate_kernel(
    const float* __restrict__ tokens, const float* __restrict__ gw,
    int* __restrict__ expert_id, float* __restrict__ tok_p,
    float* __restrict__ psum_blocks) {
  __shared__ __align__(16) float sgw[NEXP * DMODEL];  // 32 KB
  __shared__ float pshare[8][NEXP];
  for (int i = threadIdx.x; i < NEXP * DMODEL / 4; i += 256)
    ((float4*)sgw)[i] = ((const float4*)gw)[i];
  __syncthreads();

  const int lane = threadIdx.x & 31;
  const int wid = threadIdx.x >> 5;
  const int t = blockIdx.x * 8 + wid;
  const float* tok = tokens + (size_t)t * DMODEL;

  float acc[NEXP];
#pragma unroll
  for (int e = 0; e < NEXP; ++e) acc[e] = 0.0f;
  for (int i = 0; i < DMODEL / 128; ++i) {
    const int k = (i * 32 + lane) * 4;
    const float4 x = *(const float4*)(tok + k);
#pragma unroll
    for (int e = 0; e < NEXP; ++e) {
      const float4 g = *(const float4*)(sgw + e * DMODEL + k);
      acc[e] += x.x * g.x + x.y * g.y + x.z * g.z + x.w * g.w;
    }
  }
#pragma unroll
  for (int e = 0; e < NEXP; ++e)
#pragma unroll
    for (int off = 16; off > 0; off >>= 1)
      acc[e] += __shfl_xor(acc[e], off, 32);

  if (lane == 0) {
    float mx = acc[0];
    int am = 0;
#pragma unroll
    for (int e = 1; e < NEXP; ++e)
      if (acc[e] > mx) { mx = acc[e]; am = e; }
    float p[NEXP], s = 0.0f;
#pragma unroll
    for (int e = 0; e < NEXP; ++e) { p[e] = __expf(acc[e] - mx); s += p[e]; }
    const float inv = 1.0f / s;
    expert_id[t] = am;
    tok_p[t] = p[am] * inv;
#pragma unroll
    for (int e = 0; e < NEXP; ++e) pshare[wid][e] = p[e] * inv;
  }
  __syncthreads();
  if (threadIdx.x < NEXP) {
    float s = 0.0f;
#pragma unroll
    for (int wv = 0; wv < 8; ++wv) s += pshare[wv][threadIdx.x];
    psum_blocks[(size_t)blockIdx.x * NEXP + threadIdx.x] = s;
  }
}

// ---------------------------------------------------------------------------
// 2) Exact sequential ranks (cumsum semantics) via block-wide scan.
//    1024 threads x 8 tokens each. Builds slot_tok (-1 for unused slots).
// ---------------------------------------------------------------------------
__global__ __launch_bounds__(1024) void scan_kernel(
    const int* __restrict__ expert_id, int* __restrict__ slot_tok,
    int* __restrict__ fullcnt, int* __restrict__ cnt) {
  __shared__ int scn[1024][NEXP];  // 32 KB
  __shared__ int stot[NEXP];
  const int tid = threadIdx.x;
  int eids[8];
  int loc[NEXP];
#pragma unroll
  for (int e = 0; e < NEXP; ++e) loc[e] = 0;
#pragma unroll
  for (int i = 0; i < 8; ++i) {
    eids[i] = expert_id[tid * 8 + i];
    loc[eids[i]]++;
  }
  int v[NEXP];
#pragma unroll
  for (int e = 0; e < NEXP; ++e) { v[e] = loc[e]; scn[tid][e] = v[e]; }
  __syncthreads();
  for (int off = 1; off < 1024; off <<= 1) {
    int nb[NEXP];
#pragma unroll
    for (int e = 0; e < NEXP; ++e) nb[e] = 0;
    if (tid >= off) {
#pragma unroll
      for (int e = 0; e < NEXP; ++e) nb[e] = scn[tid - off][e];
    }
    __syncthreads();
    if (tid >= off) {
#pragma unroll
      for (int e = 0; e < NEXP; ++e) { v[e] += nb[e]; scn[tid][e] = v[e]; }
    }
    __syncthreads();
  }
  if (tid == 1023) {
#pragma unroll
    for (int e = 0; e < NEXP; ++e) stot[e] = v[e];
  }
  int start[NEXP];
#pragma unroll
  for (int e = 0; e < NEXP; ++e) start[e] = v[e] - loc[e];
#pragma unroll
  for (int i = 0; i < 8; ++i) {
    const int t = tid * 8 + i;
    const int e = eids[i];
    const int r = start[e]++;
    if (r < CAP) slot_tok[e * CAP + r] = t;
  }
  __syncthreads();
  if (tid < NEXP) {
    fullcnt[tid] = stot[tid];
    cnt[tid] = min(stot[tid], CAP);
  }
  for (int s = tid; s < NEXP * CAP; s += 1024) {
    const int e = s / CAP;
    const int r = s - e * CAP;
    if (r >= min(stot[e], CAP)) slot_tok[s] = -1;
  }
}

// ---------------------------------------------------------------------------
// 3) Dispatch gather: X[slot] = f16(tokens[slot_tok[slot]]) (zeros if unused)
// ---------------------------------------------------------------------------
__global__ __launch_bounds__(128) void dispatch_kernel(
    const float* __restrict__ tokens, const int* __restrict__ slot_tok,
    _Float16* __restrict__ X) {
  const int s = blockIdx.x;
  const int t = slot_tok[s];
  _Float16* xr = X + (size_t)s * DMODEL;
  const int i = threadIdx.x * 8;
  if (t < 0) {
    h8 z = {};
    *(h8*)(xr + i) = z;
  } else {
    const float* tr = tokens + (size_t)t * DMODEL;
    const float4 a = *(const float4*)(tr + i);
    const float4 b = *(const float4*)(tr + i + 4);
    h8 vv = {(_Float16)a.x, (_Float16)a.y, (_Float16)a.z, (_Float16)a.w,
             (_Float16)b.x, (_Float16)b.y, (_Float16)b.z, (_Float16)b.w};
    *(h8*)(xr + i) = vv;
  }
}

// ---------------------------------------------------------------------------
// 4) l_aux = E * sum_e mean(probs[:,e]) * (count_e / T)   (fixed-order reduce)
// ---------------------------------------------------------------------------
__global__ __launch_bounds__(256) void laux_kernel(
    const float* __restrict__ psum_blocks, const int* __restrict__ fullcnt,
    float* __restrict__ out_laux) {
  __shared__ float part[32][NEXP];
  const int e = threadIdx.x & 7;
  const int pr = threadIdx.x >> 3;  // 0..31
  float s = 0.0f;
  for (int j = 0; j < 32; ++j)
    s += psum_blocks[(size_t)(pr + 32 * j) * NEXP + e];
  part[pr][e] = s;
  __syncthreads();
  if (threadIdx.x == 0) {
    float la = 0.0f;
    for (int ee = 0; ee < NEXP; ++ee) {
      float tot = 0.0f;
      for (int p2 = 0; p2 < 32; ++p2) tot += part[p2][ee];
      la += (tot / (float)T_TOK) * ((float)fullcnt[ee] / (float)T_TOK);
    }
    *out_laux = (float)NEXP * la;
  }
}

// ---------------------------------------------------------------------------
// WMMA GEMM 1: H = f16(gelu(X @ w1 + b1)), per expert. 128x128 tiles, K=1024.
// 8 waves (2x4), each wave 64x32 via 4x2 v_wmma_f32_16x16x32_f16.
// ---------------------------------------------------------------------------
__global__ __launch_bounds__(256) void ffn1_kernel(
    const _Float16* __restrict__ X, const float* __restrict__ w1,
    const float* __restrict__ b1, const int* __restrict__ cnt,
    _Float16* __restrict__ H) {
  __shared__ __align__(16) char smem[20480];  // A(10240) + B(10240); C aliases
  _Float16* sA = (_Float16*)smem;             // [128][40] halves
  _Float16* sB = (_Float16*)(smem + 10240);   // [128][40] halves (B^T)
  _Float16* sC = (_Float16*)smem;             // [64][136] halves (epilogue)

  const int e = blockIdx.x / (CAP / 128);
  const int m0 = (blockIdx.x - e * (CAP / 128)) * 128;
  if (m0 >= cnt[e]) return;
  const int n0 = blockIdx.y * 128;

  const _Float16* Ag = X + (size_t)(e * CAP + m0) * DMODEL;
  const float* Bg = w1 + (size_t)e * DMODEL * DFF_;

  const int tid = threadIdx.x;
  const int lane = tid & 31;
  const int wm = tid >> 7;         // 0..1
  const int wn = (tid >> 5) & 3;   // 0..3

  v8f acc[4][2];
#pragma unroll
  for (int mi = 0; mi < 4; ++mi)
#pragma unroll
    for (int ni = 0; ni < 2; ++ni) acc[mi][ni] = {};

  const int arow = tid >> 2;           // 0..63
  const int apart = (tid & 3) * 8;     // halves
  const int bn = tid & 127;
  const int kb = (tid >> 7) * 2;

  h8 ra0 = *(const h8*)(Ag + (size_t)arow * DMODEL + apart);
  h8 ra1 = *(const h8*)(Ag + (size_t)(arow + 64) * DMODEL + apart);
  float rb0[8], rb1[8];
#pragma unroll
  for (int i = 0; i < 8; ++i) {
    const int k = kb + i * 4;
    rb0[i] = Bg[(size_t)k * DFF_ + n0 + bn];
    rb1[i] = Bg[(size_t)(k + 1) * DFF_ + n0 + bn];
  }
  *(h8*)(sA + arow * 40 + apart) = ra0;
  *(h8*)(sA + (arow + 64) * 40 + apart) = ra1;
#pragma unroll
  for (int i = 0; i < 8; ++i) {
    const int k = kb + i * 4;
    h2 pr = {(_Float16)rb0[i], (_Float16)rb1[i]};
    *(h2*)(sB + bn * 40 + k) = pr;
  }
  __syncthreads();

  const int KT = DMODEL / 32;
  for (int kt = 0; kt < KT; ++kt) {
    if (kt + 1 < KT) {
      const int k0 = (kt + 1) * 32;
      ra0 = *(const h8*)(Ag + (size_t)arow * DMODEL + k0 + apart);
      ra1 = *(const h8*)(Ag + (size_t)(arow + 64) * DMODEL + k0 + apart);
#pragma unroll
      for (int i = 0; i < 8; ++i) {
        const int k = kb + i * 4;
        rb0[i] = Bg[(size_t)(k0 + k) * DFF_ + n0 + bn];
        rb1[i] = Bg[(size_t)(k0 + k + 1) * DFF_ + n0 + bn];
      }
    }
    const int fr = lane & 15;
    const int kq = lane >> 4;
    v16h af[4], bf[2];
#pragma unroll
    for (int mi = 0; mi < 4; ++mi) {
      const _Float16* p = sA + (wm * 64 + mi * 16 + fr) * 40 + kq * 8;
      const h8 lo = *(const h8*)p;
      const h8 hi = *(const h8*)(p + 16);
      af[mi] = __builtin_shufflevector(lo, hi, 0, 1, 2, 3, 4, 5, 6, 7, 8, 9,
                                       10, 11, 12, 13, 14, 15);
    }
#pragma unroll
    for (int ni = 0; ni < 2; ++ni) {
      const _Float16* p = sB + (wn * 32 + ni * 16 + fr) * 40 + kq * 8;
      const h8 lo = *(const h8*)p;
      const h8 hi = *(const h8*)(p + 16);
      bf[ni] = __builtin_shufflevector(lo, hi, 0, 1, 2, 3, 4, 5, 6, 7, 8, 9,
                                       10, 11, 12, 13, 14, 15);
    }
#pragma unroll
    for (int mi = 0; mi < 4; ++mi)
#pragma unroll
      for (int ni = 0; ni < 2; ++ni)
        acc[mi][ni] = __builtin_amdgcn_wmma_f32_16x16x32_f16(
            false, af[mi], false, bf[ni], (short)0, acc[mi][ni], false, false);
    __syncthreads();
    if (kt + 1 < KT) {
      *(h8*)(sA + arow * 40 + apart) = ra0;
      *(h8*)(sA + (arow + 64) * 40 + apart) = ra1;
#pragma unroll
      for (int i = 0; i < 8; ++i) {
        const int k = kb + i * 4;
        h2 pr = {(_Float16)rb0[i], (_Float16)rb1[i]};
        *(h2*)(sB + bn * 40 + k) = pr;
      }
    }
    __syncthreads();
  }

  // epilogue: bias + GELU -> f16 H, staged through LDS (2 halves of 64 rows)
  const int fr2 = lane & 15;
  const int rbase = (lane >> 4) * 8;
  for (int half = 0; half < 2; ++half) {
    if (wm == half) {
#pragma unroll
      for (int ni = 0; ni < 2; ++ni) {
        const int cn = wn * 32 + ni * 16 + fr2;
        const float bias = b1[(size_t)e * DFF_ + n0 + cn];
#pragma unroll
        for (int mi = 0; mi < 4; ++mi) {
          const int lm = mi * 16 + rbase;
#pragma unroll
          for (int ri = 0; ri < 8; ++ri) {
            const float xv = acc[mi][ni][ri] + bias;
            sC[(lm + ri) * 136 + cn] = (_Float16)gelu_tanh(xv);
          }
        }
      }
    }
    __syncthreads();
    const int row = tid >> 2;
    const int oc = (tid & 3) * 32;
    _Float16* Hp =
        H + (size_t)(e * CAP + m0 + half * 64 + row) * DFF_ + n0 + oc;
#pragma unroll
    for (int i = 0; i < 4; ++i)
      *(h8*)(Hp + i * 8) = *(const h8*)(sC + row * 136 + oc + i * 8);
    __syncthreads();
  }
}

// ---------------------------------------------------------------------------
// WMMA GEMM 2: out[token] = tok_p * (H @ w2 + b2), fused combine scatter.
// ---------------------------------------------------------------------------
__global__ __launch_bounds__(256) void ffn2_kernel(
    const _Float16* __restrict__ H, const float* __restrict__ w2,
    const float* __restrict__ b2, const int* __restrict__ cnt,
    const int* __restrict__ slot_tok, const float* __restrict__ tok_p,
    float* __restrict__ out) {
  __shared__ __align__(16) char smem[33792];  // max(A+B=20480, C=64*132*4)
  _Float16* sA = (_Float16*)smem;
  _Float16* sB = (_Float16*)(smem + 10240);
  float* sC = (float*)smem;

  const int e = blockIdx.x / (CAP / 128);
  const int m0 = (blockIdx.x - e * (CAP / 128)) * 128;
  if (m0 >= cnt[e]) return;
  const int n0 = blockIdx.y * 128;

  const _Float16* Ag = H + (size_t)(e * CAP + m0) * DFF_;
  const float* Bg = w2 + (size_t)e * DFF_ * DMODEL;

  const int tid = threadIdx.x;
  const int lane = tid & 31;
  const int wm = tid >> 7;
  const int wn = (tid >> 5) & 3;

  v8f acc[4][2];
#pragma unroll
  for (int mi = 0; mi < 4; ++mi)
#pragma unroll
    for (int ni = 0; ni < 2; ++ni) acc[mi][ni] = {};

  const int arow = tid >> 2;
  const int apart = (tid & 3) * 8;
  const int bn = tid & 127;
  const int kb = (tid >> 7) * 2;

  h8 ra0 = *(const h8*)(Ag + (size_t)arow * DFF_ + apart);
  h8 ra1 = *(const h8*)(Ag + (size_t)(arow + 64) * DFF_ + apart);
  float rb0[8], rb1[8];
#pragma unroll
  for (int i = 0; i < 8; ++i) {
    const int k = kb + i * 4;
    rb0[i] = Bg[(size_t)k * DMODEL + n0 + bn];
    rb1[i] = Bg[(size_t)(k + 1) * DMODEL + n0 + bn];
  }
  *(h8*)(sA + arow * 40 + apart) = ra0;
  *(h8*)(sA + (arow + 64) * 40 + apart) = ra1;
#pragma unroll
  for (int i = 0; i < 8; ++i) {
    const int k = kb + i * 4;
    h2 pr = {(_Float16)rb0[i], (_Float16)rb1[i]};
    *(h2*)(sB + bn * 40 + k) = pr;
  }
  __syncthreads();

  const int KT = DFF_ / 32;
  for (int kt = 0; kt < KT; ++kt) {
    if (kt + 1 < KT) {
      const int k0 = (kt + 1) * 32;
      ra0 = *(const h8*)(Ag + (size_t)arow * DFF_ + k0 + apart);
      ra1 = *(const h8*)(Ag + (size_t)(arow + 64) * DFF_ + k0 + apart);
#pragma unroll
      for (int i = 0; i < 8; ++i) {
        const int k = kb + i * 4;
        rb0[i] = Bg[(size_t)(k0 + k) * DMODEL + n0 + bn];
        rb1[i] = Bg[(size_t)(k0 + k + 1) * DMODEL + n0 + bn];
      }
    }
    const int fr = lane & 15;
    const int kq = lane >> 4;
    v16h af[4], bf[2];
#pragma unroll
    for (int mi = 0; mi < 4; ++mi) {
      const _Float16* p = sA + (wm * 64 + mi * 16 + fr) * 40 + kq * 8;
      const h8 lo = *(const h8*)p;
      const h8 hi = *(const h8*)(p + 16);
      af[mi] = __builtin_shufflevector(lo, hi, 0, 1, 2, 3, 4, 5, 6, 7, 8, 9,
                                       10, 11, 12, 13, 14, 15);
    }
#pragma unroll
    for (int ni = 0; ni < 2; ++ni) {
      const _Float16* p = sB + (wn * 32 + ni * 16 + fr) * 40 + kq * 8;
      const h8 lo = *(const h8*)p;
      const h8 hi = *(const h8*)(p + 16);
      bf[ni] = __builtin_shufflevector(lo, hi, 0, 1, 2, 3, 4, 5, 6, 7, 8, 9,
                                       10, 11, 12, 13, 14, 15);
    }
#pragma unroll
    for (int mi = 0; mi < 4; ++mi)
#pragma unroll
      for (int ni = 0; ni < 2; ++ni)
        acc[mi][ni] = __builtin_amdgcn_wmma_f32_16x16x32_f16(
            false, af[mi], false, bf[ni], (short)0, acc[mi][ni], false, false);
    __syncthreads();
    if (kt + 1 < KT) {
      *(h8*)(sA + arow * 40 + apart) = ra0;
      *(h8*)(sA + (arow + 64) * 40 + apart) = ra1;
#pragma unroll
      for (int i = 0; i < 8; ++i) {
        const int k = kb + i * 4;
        h2 pr = {(_Float16)rb0[i], (_Float16)rb1[i]};
        *(h2*)(sB + bn * 40 + k) = pr;
      }
    }
    __syncthreads();
  }

  // epilogue: stage fp32 tile, then out[t] = tok_p[t] * (acc + b2)  (scatter)
  const int fr2 = lane & 15;
  const int rbase = (lane >> 4) * 8;
  for (int half = 0; half < 2; ++half) {
    if (wm == half) {
#pragma unroll
      for (int ni = 0; ni < 2; ++ni) {
        const int cn = wn * 32 + ni * 16 + fr2;
#pragma unroll
        for (int mi = 0; mi < 4; ++mi) {
          const int lm = mi * 16 + rbase;
#pragma unroll
          for (int ri = 0; ri < 8; ++ri)
            sC[(lm + ri) * 132 + cn] = acc[mi][ni][ri];
        }
      }
    }
    __syncthreads();
    const int row = tid >> 2;
    const int oc = (tid & 3) * 32;
    const int t = slot_tok[e * CAP + m0 + half * 64 + row];
    if (t >= 0) {
      const float w = tok_p[t];
      float* op = out + (size_t)t * DMODEL + n0 + oc;
      const float* bp = b2 + (size_t)e * DMODEL + n0 + oc;
#pragma unroll
      for (int i = 0; i < 8; ++i) {
        f4v c = *(const f4v*)(sC + row * 132 + oc + i * 4);
        f4v bb = *(const f4v*)(bp + i * 4);
        f4v r;
        r[0] = w * (c[0] + bb[0]);
        r[1] = w * (c[1] + bb[1]);
        r[2] = w * (c[2] + bb[2]);
        r[3] = w * (c[3] + bb[3]);
        *(f4v*)(op + i * 4) = r;
      }
    }
    __syncthreads();
  }
}

// ---------------------------------------------------------------------------
// Workspace layout (bytes): needs ~105 MB
//   [0,32)        fullcnt[8]
//   [32,64)       cnt[8]
//   [64,32832)    psum_blocks[1024*8] f32
//   [32896,..)    expert_id[T] i32
//   [+4T]         tok_p[T] f32
//   [+8T]         slot_tok[E*CAP] i32
//   [139392,..)   X f16 [E*CAP][D]   (20.97 MB)
//   [+X]          H f16 [E*CAP][DFF] (83.9 MB)
// ---------------------------------------------------------------------------
extern "C" void kernel_launch(void* const* d_in, const int* in_sizes, int n_in,
                              void* d_out, int out_size, void* d_ws,
                              size_t ws_size, hipStream_t stream) {
  (void)in_sizes; (void)n_in; (void)ws_size;
  const float* tokens = (const float*)d_in[0];
  const float* gw = (const float*)d_in[1];
  const float* w1 = (const float*)d_in[2];
  const float* b1 = (const float*)d_in[3];
  const float* w2 = (const float*)d_in[4];
  const float* b2 = (const float*)d_in[5];
  float* out = (float*)d_out;

  char* ws = (char*)d_ws;
  int* fullcnt = (int*)(ws + 0);
  int* cnt = (int*)(ws + 32);
  float* psum_blocks = (float*)(ws + 64);
  int* expert_id = (int*)(ws + 32896);
  float* tok_p = (float*)(ws + 32896 + 4 * T_TOK);
  int* slot_tok = (int*)(ws + 32896 + 8 * T_TOK);
  _Float16* X = (_Float16*)(ws + 139392);
  _Float16* H =
      (_Float16*)(ws + 139392 + (size_t)2 * NEXP * CAP * DMODEL);

  hipMemsetAsync(d_out, 0, (size_t)out_size * sizeof(float), stream);
  gate_kernel<<<T_TOK / 8, 256, 0, stream>>>(tokens, gw, expert_id, tok_p,
                                             psum_blocks);
  scan_kernel<<<1, 1024, 0, stream>>>(expert_id, slot_tok, fullcnt, cnt);
  dispatch_kernel<<<NEXP * CAP, 128, 0, stream>>>(tokens, slot_tok, X);
  laux_kernel<<<1, 256, 0, stream>>>(psum_blocks, fullcnt,
                                     out + (size_t)T_TOK * DMODEL);
  ffn1_kernel<<<dim3(NEXP * (CAP / 128), DFF_ / 128), 256, 0, stream>>>(
      X, w1, b1, cnt, H);
  ffn2_kernel<<<dim3(NEXP * (CAP / 128), DMODEL / 128), 256, 0, stream>>>(
      H, w2, b2, cnt, slot_tok, tok_p, out);
}
